// Policy_68195490726227
// MI455X (gfx1250) — compile-verified, matrix operand order checked by reading, not run on hardware
//
#include <hip/hip_runtime.h>
#include <hip/hip_bf16.h>
#include <cmath>

// ---------- vector types ----------
typedef __attribute__((ext_vector_type(16))) __bf16          v16bf;
typedef __attribute__((ext_vector_type(16))) unsigned short  v16us;
typedef __attribute__((ext_vector_type(8)))  unsigned short  v8us;
typedef __attribute__((ext_vector_type(8)))  float           v8f;

#define N_USER 2048
#define N_BS   1024
#define N_H    1024
#define N_S    2049
#define N_K2   2048   // 2*H
#define LN_EPSF 1e-5f

// ---------- helpers ----------
static __device__ __forceinline__ unsigned short f2bf(float f) {
    unsigned u = __builtin_bit_cast(unsigned, f);
    unsigned r = (u + 0x7FFFu + ((u >> 16) & 1u)) >> 16;   // RNE
    return (unsigned short)r;
}
static __device__ __forceinline__ float sigf(float x) { return 1.0f / (1.0f + expf(-x)); }

static __device__ __forceinline__ float blockReduceSum(float v) {
    __shared__ float red[9];
    __syncthreads();
    for (int o = 16; o; o >>= 1) v += __shfl_xor(v, o, 32);
    if ((threadIdx.x & 31) == 0) red[threadIdx.x >> 5] = v;
    __syncthreads();
    if (threadIdx.x == 0) {
        float s = 0.f; int nw = (blockDim.x + 31) >> 5;
        for (int i = 0; i < nw; ++i) s += red[i];
        red[8] = s;
    }
    __syncthreads();
    return red[8];
}
static __device__ __forceinline__ float blockReduceMax(float v) {
    __shared__ float redm[9];
    __syncthreads();
    for (int o = 16; o; o >>= 1) v = fmaxf(v, __shfl_xor(v, o, 32));
    if ((threadIdx.x & 31) == 0) redm[threadIdx.x >> 5] = v;
    __syncthreads();
    if (threadIdx.x == 0) {
        float s = -3.4e38f; int nw = (blockDim.x + 31) >> 5;
        for (int i = 0; i < nw; ++i) s = fmaxf(s, redm[i]);
        redm[8] = s;
    }
    __syncthreads();
    return redm[8];
}

static __device__ __forceinline__ v8f wmma_bf16(v16us a, v16us b, v8f c) {
    return __builtin_amdgcn_wmma_f32_16x16x32_bf16(false, __builtin_bit_cast(v16bf, a),
                                                   false, __builtin_bit_cast(v16bf, b),
                                                   (short)0, c, false, false);
}

// ---------- preamble kernels ----------
__global__ void init_kernel(const int* __restrict__ Action, float* mask, float* out, float* scal) {
    int i = blockIdx.x * 256 + threadIdx.x;
    if (i < N_S)  mask[i] = 1.0f;
    if (i < 16)   out[i]  = (float)(Action[i] - 1);   // schedule_result
    if (i == 16)  out[16] = 0.0f;                     // logp accumulator
    if (i >= 20 && i < 24) scal[i - 20] = 0.0f;       // emb/key sum accumulators
}

__global__ void tobf16_kernel(const float* __restrict__ x, unsigned short* __restrict__ y, long n) {
    long i = (long)blockIdx.x * 256 + threadIdx.x;
    if (i < n) y[i] = f2bf(x[i]);
}

// conv 3x3 over (3, 2050, 1026) -> bf16 (2048, 1024), input pre-scaled by 1e7
__global__ void conv_kernel(const float* __restrict__ inp, const float* __restrict__ cw,
                            const float* __restrict__ cb, unsigned short* __restrict__ convb) {
    int idx = blockIdx.x * 256 + threadIdx.x;
    if (idx >= N_USER * N_BS) return;
    int s = idx >> 10, j = idx & 1023;
    float acc = cb[0];
#pragma unroll
    for (int c = 0; c < 3; ++c)
#pragma unroll
        for (int kh = 0; kh < 3; ++kh)
#pragma unroll
            for (int kw = 0; kw < 3; ++kw)
                acc += 1.0e7f * inp[((c * 2050) + (s + kh)) * 1026 + (j + kw)] * cw[(c * 3 + kh) * 3 + kw];
    convb[idx] = f2bf(acc);
}

// generic GEMV: y[row] = W[row,:] . x + b[row], one wave per row
__global__ __launch_bounds__(256) void gemv_kernel(const float* __restrict__ W, const float* __restrict__ x,
                                                   const float* __restrict__ b, float* __restrict__ y,
                                                   int M, int K) {
    int row = blockIdx.x * 8 + (threadIdx.x >> 5);
    int lane = threadIdx.x & 31;
    if (row >= M) return;
    const float* wr = W + (size_t)row * K;
    float s = 0.f;
    for (int k = lane; k < K; k += 32) s += wr[k] * x[k];
    for (int o = 16; o; o >>= 1) s += __shfl_xor(s, o, 32);
    if (lane == 0) y[row] = s + b[row];
}

// WMMA GEMM: emb[s,h] = sigmoid( conv[s,:].bf16 . aff_w[h,:].bf16 + aff_b[h] )
__global__ __launch_bounds__(256) void emb_gemm_kernel(const unsigned short* __restrict__ convb,
                                                       const unsigned short* __restrict__ affb,
                                                       const float* __restrict__ aff_b,
                                                       float* __restrict__ emb) {
    int s0 = blockIdx.x * 16;
    int hb = blockIdx.y * 128 + (threadIdx.x >> 5) * 16;   // wave's 16 h-columns
    int lane = threadIdx.x & 31, lidx = lane & 15, hs = lane >> 4, kb = hs * 8;
    const unsigned short* arow = convb + (size_t)(s0 + lidx) * N_BS;
    const unsigned short* brow = affb + (size_t)(hb + lidx) * N_BS;
    v8f acc = {};
    for (int k0 = 0; k0 < N_BS; k0 += 32) {
        v16us au, bu;
        au.lo = *(const v8us*)(arow + k0 + kb);
        au.hi = *(const v8us*)(arow + k0 + kb + 16);
        bu.lo = *(const v8us*)(brow + k0 + kb);
        bu.hi = *(const v8us*)(brow + k0 + kb + 16);
        acc = wmma_bf16(au, bu, acc);
    }
    int n = hb + lidx;
    float bias = aff_b[n];
#pragma unroll
    for (int r = 0; r < 8; ++r) {
        int m = r + hs * 8;
        emb[(size_t)(s0 + m) * N_H + n] = sigf(acc[r] + bias);
    }
}

// sum + sumsq reduction into acc[0], acc[1] (atomic)
__global__ void reduce_sums_kernel(const float* __restrict__ x, long n, float* acc) {
    float s = 0.f, q = 0.f;
    for (long i = (long)blockIdx.x * blockDim.x + threadIdx.x; i < n; i += (long)gridDim.x * blockDim.x) {
        float v = x[i]; s += v; q += v * v;
    }
    s = blockReduceSum(s); q = blockReduceSum(q);
    if (threadIdx.x == 0) { atomicAdd(&acc[0], s); atomicAdd(&acc[1], q); }
}

// gin0[h] = mean over s of emb[s,h]
__global__ void colmean_kernel(const float* __restrict__ emb, float* __restrict__ gin) {
    int h = blockIdx.x * 256 + threadIdx.x;
    if (h >= N_H) return;
    float s = 0.f;
    for (int r = 0; r < N_USER; ++r) s += emb[(size_t)r * N_H + h];
    gin[h] = s * (1.0f / (float)N_USER);
}

// key_m = ln2d([zeros; emb]) with scalar mean/var; also accumulate key sum/sumsq
__global__ void keym_kernel(const float* __restrict__ emb, const float* __restrict__ lnw,
                            const float* __restrict__ lnb, const float* __restrict__ scal,
                            float* __restrict__ key_m, float* keyacc) {
    long idx = (long)blockIdx.x * 256 + threadIdx.x;     // over 2049*1024, exact grid
    int s = (int)(idx >> 10), k = (int)(idx & 1023);
    const float NT = (float)N_S * (float)N_H;
    float m = scal[0] / NT;
    float var = scal[1] / NT - m * m;
    float inv = rsqrtf(var + LN_EPSF);
    float x = (s == 0) ? 0.f : emb[(size_t)(s - 1) * N_H + k];
    float val = (x - m) * inv * lnw[idx] + lnb[idx];
    key_m[idx] = val;
    float s1 = blockReduceSum(val);
    float s2 = blockReduceSum(val * val);
    if (threadIdx.x == 0) { atomicAdd(&keyacc[0], s1); atomicAdd(&keyacc[1], s2); }
}

// ---------- per-step kernels ----------
// scalar LN stats for cat = [key_m | bcast ctx]; also zero attn output / context accumulators
__global__ void stats_zero_kernel(const float* __restrict__ ctx, const float* __restrict__ keyacc,
                                  float* statout, float* ztarget, int zn, float* ztarget2, int zn2) {
    float s = 0.f, q = 0.f;
    for (int i = threadIdx.x; i < N_H; i += 256) { float v = ctx[i]; s += v; q += v * v; }
    s = blockReduceSum(s); q = blockReduceSum(q);
    if (threadIdx.x == 0) {
        float tot  = keyacc[0] + (float)N_S * s;
        float totq = keyacc[1] + (float)N_S * q;
        const float NT = (float)N_S * (float)N_K2;
        float m = tot / NT;
        float var = totq / NT - m * m;
        statout[0] = m;
        statout[1] = rsqrtf(var + LN_EPSF);
    }
    for (int i = threadIdx.x; i < zn; i += 256) ztarget[i] = 0.f;
    for (int i = threadIdx.x; i < zn2; i += 256) ztarget2[i] = 0.f;
}

// materialize normalized cat in bf16: (2049 x 2048)
__global__ void catprep_kernel(const float* __restrict__ key_m, const float* __restrict__ ctx,
                               const float* __restrict__ lnw, const float* __restrict__ lnb,
                               const float* __restrict__ stats, unsigned short* __restrict__ catb) {
    long idx = (long)blockIdx.x * 256 + threadIdx.x;     // over 2049*2048, exact grid
    int s = (int)(idx >> 11), k = (int)(idx & 2047);
    float m = stats[0], inv = stats[1];
    float x = (k < N_H) ? key_m[(size_t)s * N_H + k] : ctx[k - N_H];
    catb[idx] = f2bf((x - m) * inv * lnw[idx] + lnb[idx]);
    (void)s;
}

// fused WMMA attn, 2x2 register-blocked: each wave computes a 32(w) x 32(s) tile.
// out[s] += sum_w V[w] * tanh( W[w,:] . cat[s,:] )
__global__ __launch_bounds__(256) void attn_gemm_kernel(const unsigned short* __restrict__ Wb,
                                                        const float* __restrict__ V,
                                                        const unsigned short* __restrict__ catb,
                                                        float* __restrict__ out) {
    int wave = threadIdx.x >> 5;
    int ww = wave & 3, wsv = wave >> 2;                 // 4 (w) x 2 (s) wave grid
    int w0 = blockIdx.y * 128 + ww * 32;
    int s0 = blockIdx.x * 64 + wsv * 32;
    int lane = threadIdx.x & 31, lidx = lane & 15, hs = lane >> 4, kb = hs * 8;
    const unsigned short* arow0 = Wb + (size_t)(w0 + lidx) * N_K2;
    const unsigned short* arow1 = Wb + (size_t)(w0 + 16 + lidx) * N_K2;
    int sr0 = s0 + lidx;      if (sr0 >= N_S) sr0 = N_S - 1;   // tail clamp
    int sr1 = s0 + 16 + lidx; if (sr1 >= N_S) sr1 = N_S - 1;
    const unsigned short* brow0 = catb + (size_t)sr0 * N_K2;
    const unsigned short* brow1 = catb + (size_t)sr1 * N_K2;
    v8f acc00 = {}, acc01 = {}, acc10 = {}, acc11 = {};
    for (int k0 = 0; k0 < N_K2; k0 += 32) {
        __builtin_prefetch(arow0 + k0 + 256, 0, 1);    // global_prefetch_b8, 512B ahead
        v16us a0, a1, b0, b1;
        a0.lo = *(const v8us*)(arow0 + k0 + kb); a0.hi = *(const v8us*)(arow0 + k0 + kb + 16);
        a1.lo = *(const v8us*)(arow1 + k0 + kb); a1.hi = *(const v8us*)(arow1 + k0 + kb + 16);
        b0.lo = *(const v8us*)(brow0 + k0 + kb); b0.hi = *(const v8us*)(brow0 + k0 + kb + 16);
        b1.lo = *(const v8us*)(brow1 + k0 + kb); b1.hi = *(const v8us*)(brow1 + k0 + kb + 16);
        acc00 = wmma_bf16(a0, b0, acc00);
        acc01 = wmma_bf16(a0, b1, acc01);
        acc10 = wmma_bf16(a1, b0, acc10);
        acc11 = wmma_bf16(a1, b1, acc11);
    }
    // epilogue: fold V * tanh over the 32 w-rows this wave owns, per s-column
    float p0 = 0.f, p1 = 0.f;
#pragma unroll
    for (int r = 0; r < 8; ++r) {
        float v0 = V[w0 + r + hs * 8];
        float v1 = V[w0 + 16 + r + hs * 8];
        p0 += v0 * tanhf(acc00[r]) + v1 * tanhf(acc10[r]);
        p1 += v0 * tanhf(acc01[r]) + v1 * tanhf(acc11[r]);
    }
    p0 += __shfl_xor(p0, 16, 32);        // combine half-lanes holding the same column
    p1 += __shfl_xor(p1, 16, 32);
    if (hs == 0) {
        if (s0 + lidx < N_S)      atomicAdd(&out[s0 + lidx], p0);
        if (s0 + 16 + lidx < N_S) atomicAdd(&out[s0 + 16 + lidx], p1);
    }
}

// c[k] += sum_{s in chunk} key_m[s,k] * u[s]   (c pre-zeroed by stats_zero)
__global__ void keymT_mv_kernel(const float* __restrict__ key_m, const float* __restrict__ u,
                                float* __restrict__ c) {
    int k = blockIdx.x * 256 + threadIdx.x;
    if (k >= N_H) return;
    int sBeg = blockIdx.y * 128;
    int sEnd = sBeg + 128; if (sEnd > N_S) sEnd = N_S;
    float s = 0.f;
    for (int r = sBeg; r < sEnd; ++r) s += key_m[(size_t)r * N_H + k] * u[r];
    atomicAdd(&c[k], s);
}

// softmax(logits*mask), logp accumulation, mask update
__global__ void finalize_kernel(const float* __restrict__ logits, float* mask,
                                const int* __restrict__ Action, int step, float* out) {
    float mx = -3.4e38f;
    for (int i = threadIdx.x; i < N_S; i += 256) mx = fmaxf(mx, logits[i] * mask[i]);
    mx = blockReduceMax(mx);
    float s = 0.f;
    for (int i = threadIdx.x; i < N_S; i += 256) s += expf(logits[i] * mask[i] - mx);
    s = blockReduceSum(s);
    if (threadIdx.x == 0) {
        int sel = Action[step];
        out[16] += logits[sel] * mask[sel] - mx - logf(s);
        mask[sel] = 1.0e-6f;
    }
}

// GRU cell update + gin_new = emb[sel-1]
__global__ void gru_update_kernel(const float* __restrict__ gi, const float* __restrict__ gh,
                                  const float* __restrict__ emb, const int* __restrict__ Action,
                                  int step, float* __restrict__ h, float* __restrict__ gin) {
    int j = blockIdx.x * 256 + threadIdx.x;
    if (j >= N_H) return;
    float r = sigf(gi[j] + gh[j]);
    float z = sigf(gi[N_H + j] + gh[N_H + j]);
    float n = tanhf(gi[2 * N_H + j] + r * gh[2 * N_H + j]);
    h[j] = (1.f - z) * n + z * h[j];
    int sel = Action[step];
    int row = sel - 1; if (row < 0) row += N_USER;   // python negative index
    gin[j] = emb[(size_t)row * N_H + j];
}

// ---------- host launch ----------
extern "C" void kernel_launch(void* const* d_in, const int* in_sizes, int n_in,
                              void* d_out, int out_size, void* d_ws, size_t ws_size,
                              hipStream_t stream) {
    const float* input    = (const float*)d_in[0];
    const float* avg_rew  = (const float*)d_in[1];
    const float* conv_w   = (const float*)d_in[2];
    const float* conv_b   = (const float*)d_in[3];
    const float* aff_w    = (const float*)d_in[4];
    const float* aff_b    = (const float*)d_in[5];
    const float* rew_w    = (const float*)d_in[6];
    const float* rew_b    = (const float*)d_in[7];
    const float* W_a      = (const float*)d_in[8];
    const float* V_a      = (const float*)d_in[9];
    const float* W_c      = (const float*)d_in[10];
    const float* V_c      = (const float*)d_in[11];
    const float* ln_w     = (const float*)d_in[12];
    const float* ln_b     = (const float*)d_in[13];
    const float* ln1_w    = (const float*)d_in[14];
    const float* ln1_b    = (const float*)d_in[15];
    const float* ln2_w    = (const float*)d_in[16];
    const float* ln2_b    = (const float*)d_in[17];
    const float* gru_wih  = (const float*)d_in[18];
    const float* gru_whh  = (const float*)d_in[19];
    const float* gru_bih  = (const float*)d_in[20];
    const float* gru_bhh  = (const float*)d_in[21];
    const int*   Action   = (const int*)d_in[22];
    float* out = (float*)d_out;

    // workspace carve-up (~40 MB)
    char* base = (char*)d_ws;
    auto alloc = [&](size_t bytes) -> void* {
        void* p = (void*)base;
        base += (bytes + 255) & ~(size_t)255;
        return p;
    };
    float* emb    = (float*)alloc((size_t)N_USER * N_H * 4);
    float* key_m  = (float*)alloc((size_t)N_S * N_H * 4);
    unsigned short* convb = (unsigned short*)alloc((size_t)N_USER * N_BS * 2);
    unsigned short* affb  = (unsigned short*)alloc((size_t)N_H * N_BS * 2);
    unsigned short* Wab   = (unsigned short*)alloc((size_t)N_H * N_K2 * 2);
    unsigned short* Wcb   = (unsigned short*)alloc((size_t)N_H * N_K2 * 2);
    unsigned short* catb  = (unsigned short*)alloc((size_t)N_S * N_K2 * 2);
    float* u      = (float*)alloc(2064 * 4);
    float* logits = (float*)alloc(2064 * 4);
    float* mask   = (float*)alloc(2064 * 4);
    float* h      = (float*)alloc(1024 * 4);
    float* gin    = (float*)alloc(1024 * 4);
    float* cvec   = (float*)alloc(1024 * 4);
    float* gi     = (float*)alloc(3072 * 4);
    float* gh     = (float*)alloc(3072 * 4);
    float* scal   = (float*)alloc(64 * 4);   // [0,1]=emb sums  [2,3]=key sums  [4,5]/[6,7]=LN stats

    // ---- preamble ----
    init_kernel<<<9, 256, 0, stream>>>(Action, mask, out, scal);
    tobf16_kernel<<<(1024 * 1024 + 255) / 256, 256, 0, stream>>>(aff_w, affb, 1024L * 1024);
    tobf16_kernel<<<(1024 * 2048 + 255) / 256, 256, 0, stream>>>(W_a, Wab, 1024L * 2048);
    tobf16_kernel<<<(1024 * 2048 + 255) / 256, 256, 0, stream>>>(W_c, Wcb, 1024L * 2048);
    conv_kernel<<<(N_USER * N_BS) / 256, 256, 0, stream>>>(input, conv_w, conv_b, convb);
    gemv_kernel<<<128, 256, 0, stream>>>(rew_w, avg_rew, rew_b, h, 1024, 2048);   // h0
    emb_gemm_kernel<<<dim3(N_USER / 16, N_H / 128), 256, 0, stream>>>(convb, affb, aff_b, emb);
    reduce_sums_kernel<<<256, 256, 0, stream>>>(emb, (long)N_USER * N_H, scal);
    colmean_kernel<<<4, 256, 0, stream>>>(emb, gin);
    keym_kernel<<<(N_S * N_H) / 256, 256, 0, stream>>>(emb, ln_w, ln_b, scal, key_m, scal + 2);

    // ---- 17-step scan ----
    for (int step = 0; step < 17; ++step) {
        // head a: ctx = h   (also zero u and cvec accumulators)
        stats_zero_kernel<<<1, 256, 0, stream>>>(h, scal + 2, scal + 4, u, N_S, cvec, N_H);
        catprep_kernel<<<(N_S * N_K2) / 256, 256, 0, stream>>>(key_m, h, ln1_w, ln1_b, scal + 4, catb);
        attn_gemm_kernel<<<dim3(33, 8), 256, 0, stream>>>(Wab, V_a, catb, u);
        // c = key_m^T . u  (s-range split across grid.y, atomic accumulate)
        keymT_mv_kernel<<<dim3(4, 17), 256, 0, stream>>>(key_m, u, cvec);
        // head c: ctx = c
        stats_zero_kernel<<<1, 256, 0, stream>>>(cvec, scal + 2, scal + 6, logits, N_S, nullptr, 0);
        catprep_kernel<<<(N_S * N_K2) / 256, 256, 0, stream>>>(key_m, cvec, ln2_w, ln2_b, scal + 6, catb);
        attn_gemm_kernel<<<dim3(33, 8), 256, 0, stream>>>(Wcb, V_c, catb, logits);
        // GRU gate GEMVs (use old gin / old h)
        gemv_kernel<<<384, 256, 0, stream>>>(gru_wih, gin, gru_bih, gi, 3072, 1024);
        gemv_kernel<<<384, 256, 0, stream>>>(gru_whh, h, gru_bhh, gh, 3072, 1024);
        // softmax + logp + mask, then state update
        finalize_kernel<<<1, 256, 0, stream>>>(logits, mask, Action, step, out);
        gru_update_kernel<<<4, 256, 0, stream>>>(gi, gh, emb, Action, step, h, gin);
    }
    (void)in_sizes; (void)n_in; (void)out_size; (void)ws_size;
}
